// LSA_60524679135246
// MI455X (gfx1250) — matrix operand order repaired and checked by reading, not compiled, not run on hardware
//
#include <hip/hip_runtime.h>

// ---------------------------------------------------------------------------
// Local self-attention w/ spectral-normed projections (MI455X / gfx1250).
// B=2, S=2048, D=512, H=8, Dh=64, WINDOW=8 (recent causal band masked OUT).
// Matmuls: v_wmma_f32_16x16x32_bf16 (fp32 accum). K/V tiles staged into LDS
// by the Tensor Data Mover (tensor_load_to_lds, TENSORcnt double-buffering).
// ---------------------------------------------------------------------------

typedef __attribute__((ext_vector_type(16))) __bf16        v16bf;
typedef __attribute__((ext_vector_type(8)))  float         v8f;
typedef __attribute__((ext_vector_type(8)))  unsigned int  v8u;
typedef __attribute__((ext_vector_type(4)))  unsigned int  u32x4;
typedef __attribute__((ext_vector_type(8)))  unsigned int  u32x8;

#define S_LEN   2048
#define INNER   512
#define QKV_LD  1536

// fp32 -> bf16 round-to-nearest-even (bits)
__device__ __forceinline__ unsigned short f2bf(float f) {
  unsigned int u = __float_as_uint(f);
  u += 0x7FFFu + ((u >> 16) & 1u);
  return (unsigned short)(u >> 16);
}

__device__ __forceinline__ float hmax16(float x) {
  x = fmaxf(x, __shfl_xor(x, 1, 32));
  x = fmaxf(x, __shfl_xor(x, 2, 32));
  x = fmaxf(x, __shfl_xor(x, 4, 32));
  x = fmaxf(x, __shfl_xor(x, 8, 32));
  return x;
}
__device__ __forceinline__ float hsum16(float x) {
  x += __shfl_xor(x, 1, 32);
  x += __shfl_xor(x, 2, 32);
  x += __shfl_xor(x, 4, 32);
  x += __shfl_xor(x, 8, 32);
  return x;
}

__device__ __forceinline__ float block_sum(float x, float* red, int tid) {
  red[tid] = x; __syncthreads();
  for (int s = 128; s > 0; s >>= 1) {
    if (tid < s) red[tid] += red[tid + s];
    __syncthreads();
  }
  float r = red[0]; __syncthreads();
  return r;
}

// byte offset of a shared-memory object within the workgroup LDS allocation
__device__ __forceinline__ unsigned lds_off(const void* p) {
  return (unsigned)(size_t)(const __attribute__((address_space(3))) void*)p;
}

// Issue one TDM load of a 2-D bf16 tile [32 rows x 64 cols], row stride 1536
// elements, into LDS at lds_byte (row-major, densely packed). Wave-uniform.
__device__ __forceinline__ void tdm_load_tile_32x64(unsigned lds_byte,
                                                    const void* gptr) {
  unsigned long long ga = (unsigned long long)(size_t)gptr;
  u32x4 g0 = {
      1u,                                        // count=1, user descriptor
      lds_byte,                                  // lds_addr (bytes)
      (unsigned)ga,                              // global_addr[31:0]
      ((unsigned)(ga >> 32) & 0x01FFFFFFu) | 0x80000000u  // addr[56:32] | type=2
  };
  u32x8 g1 = {
      0x00010000u,   // workgroup_mask=0, data_size=1 (2 bytes)
      64u << 16,     // atomic_barrier=0 | tensor_dim0 = 64
      32u << 16,     // tensor_dim0.hi=0 | tensor_dim1 = 32
      64u << 16,     // tensor_dim1.hi=0 | tile_dim0 = 64
      32u,           // tile_dim1 = 32, tile_dim2 = 0
      1536u,         // tensor_dim0_stride = 1536 (elements)
      0u,            // stride.hi=0 | tensor_dim1_stride=0 (2-D tile)
      0u
  };
  asm volatile("tensor_load_to_lds %0, %1" :: "s"(g0), "s"(g1) : "memory");
}

// ---------------------------------------------------------------------------
// Spectral-norm scale: one power-iteration step, scale = sigma / (v^T W u2).
// ---------------------------------------------------------------------------
__global__ __launch_bounds__(256) void sn_sigma(
    const float* __restrict__ W, const float* __restrict__ u,
    const float* __restrict__ sigma, float* __restrict__ scale,
    int R, int C) {
  __shared__ float sv[1536];
  __shared__ float su[512];
  __shared__ float sred[256];
  int tid = threadIdx.x;

  for (int r = tid; r < R; r += 256) {
    const float* wr = W + (size_t)r * C;
    float acc = 0.f;
    for (int c = 0; c < C; ++c) acc += wr[c] * u[c];
    sv[r] = acc;
  }
  __syncthreads();

  float loc = 0.f;
  for (int r = tid; r < R; r += 256) loc += sv[r] * sv[r];
  float vn = rsqrtf(block_sum(loc, sred, tid));

  for (int c = tid; c < C; c += 256) {
    float acc = 0.f;
    for (int r = 0; r < R; ++r) acc += W[(size_t)r * C + c] * sv[r];
    su[c] = acc * vn;
  }
  __syncthreads();

  loc = 0.f;
  for (int c = tid; c < C; c += 256) loc += su[c] * su[c];
  float un = rsqrtf(block_sum(loc, sred, tid));

  loc = 0.f;
  for (int r = tid; r < R; r += 256) {
    const float* wr = W + (size_t)r * C;
    float acc = 0.f;
    for (int c = 0; c < C; ++c) acc += wr[c] * su[c];
    loc += (sv[r] * vn) * (acc * un);
  }
  float sigma_w = block_sum(loc, sred, tid);
  if (tid == 0) scale[0] = sigma[0] / sigma_w;
}

// ---------------------------------------------------------------------------
// GEMM  Y[m,n] = (sum_k A[m,k] * W[n,k]) * scale   (fp32 in, bf16 out)
// ---------------------------------------------------------------------------
__global__ __launch_bounds__(32) void gemm_f32_bf16out(
    const float* __restrict__ A, const float* __restrict__ W,
    const float* __restrict__ scale, unsigned short* __restrict__ Y,
    int K, int N) {
  int lane = threadIdx.x & 31;
  int g = lane >> 4, ln = lane & 15;
  int n0 = blockIdx.x << 4, m0 = blockIdx.y << 4;
  const float* arow = A + (size_t)(m0 + ln) * K;
  const float* wrow = W + (size_t)(n0 + ln) * K;
  v8f acc = {};
  for (int k0 = 0; k0 < K; k0 += 32) {
    v8u ap, bp;
#pragma unroll
    for (int v = 0; v < 8; ++v) {
      int Ka = (g << 4) + (v << 1);                               // A contiguous
      float2 fa = *(const float2*)(arow + k0 + Ka);
      ap[v] = (unsigned int)f2bf(fa.x) | ((unsigned int)f2bf(fa.y) << 16);
      int Kb = (g << 4) + (v << 1);                               // B contiguous
      float2 fb = *(const float2*)(wrow + k0 + Kb);
      bp[v] = (unsigned int)f2bf(fb.x) | ((unsigned int)f2bf(fb.y) << 16);
    }
    acc = __builtin_amdgcn_wmma_f32_16x16x32_bf16(
        false, __builtin_bit_cast(v16bf, ap),
        false, __builtin_bit_cast(v16bf, bp),
        (short)0, acc, false, false);
  }
  float sc = scale[0];
#pragma unroll
  for (int r = 0; r < 8; ++r) {
    int m = m0 + r + (g << 3);
    Y[(size_t)m * N + n0 + ln] = f2bf(acc[r] * sc);
  }
}

// ---------------------------------------------------------------------------
// GEMM  Y[m,n] = (sum_k A[m,k] * W[n,k]) * scale + bias[n]  (bf16 A, fp32 out)
// ---------------------------------------------------------------------------
__global__ __launch_bounds__(32) void gemm_bf16_f32out(
    const unsigned short* __restrict__ A, const float* __restrict__ W,
    const float* __restrict__ bias, const float* __restrict__ scale,
    float* __restrict__ Y, int K, int N) {
  int lane = threadIdx.x & 31;
  int g = lane >> 4, ln = lane & 15;
  int n0 = blockIdx.x << 4, m0 = blockIdx.y << 4;
  const unsigned short* arow = A + (size_t)(m0 + ln) * K;
  const float* wrow = W + (size_t)(n0 + ln) * K;
  v8f acc = {};
  for (int k0 = 0; k0 < K; k0 += 32) {
    v8u ap, bp;
#pragma unroll
    for (int v = 0; v < 8; ++v) {
      int Ka = (g << 4) + (v << 1);
      ap[v] = *(const unsigned int*)(arow + k0 + Ka);
      int Kb = (g << 4) + (v << 1);
      float2 fb = *(const float2*)(wrow + k0 + Kb);
      bp[v] = (unsigned int)f2bf(fb.x) | ((unsigned int)f2bf(fb.y) << 16);
    }
    acc = __builtin_amdgcn_wmma_f32_16x16x32_bf16(
        false, __builtin_bit_cast(v16bf, ap),
        false, __builtin_bit_cast(v16bf, bp),
        (short)0, acc, false, false);
  }
  float sc = scale[0];
#pragma unroll
  for (int r = 0; r < 8; ++r) {
    int m = m0 + r + (g << 3);
    Y[(size_t)m * N + n0 + ln] = acc[r] * sc + bias[n0 + ln];
  }
}

// ---------------------------------------------------------------------------
// Flash-style attention. One wave per (b,h, 16-query tile).
// K/V 32x64 bf16 tiles DMA'd into LDS by the TDM, double-buffered on
// TENSORcnt. dots = (Q K^T)*exp(temperature); band j in [i-7,i] -> -FLT_MAX;
// streaming softmax; O += P V via WMMA (P transposed C->A layout via LDS).
// ---------------------------------------------------------------------------
__global__ __launch_bounds__(32) void lsa_attn(
    const unsigned short* __restrict__ qkv, const float* __restrict__ temp,
    unsigned short* __restrict__ attn) {
  __shared__ __align__(16) unsigned short Kt[2][32 * 64];
  __shared__ __align__(16) unsigned short Vt[2][32 * 64];
  __shared__ unsigned short Plds[16 * 32];

  int lane = threadIdx.x & 31;
  int g = lane >> 4, ln = lane & 15;
  int bh = blockIdx.y, b = bh >> 3, h = bh & 7;
  int i0 = blockIdx.x << 4;

  const unsigned short* base = qkv + (size_t)b * S_LEN * QKV_LD + h * 64;
  const unsigned short* kg = base + 512;
  const unsigned short* vg = base + 1024;

  // Q A-fragments for the whole tile (Dh=64 -> two K=32 fragments)
  const unsigned short* qrow = base + (size_t)(i0 + ln) * QKV_LD;
  v8u aq0p, aq1p;
#pragma unroll
  for (int v = 0; v < 8; ++v) {
    int Ka = (g << 4) + (v << 1);
    aq0p[v] = *(const unsigned int*)(qrow + Ka);
    aq1p[v] = *(const unsigned int*)(qrow + 32 + Ka);
  }
  v16bf aq0 = __builtin_bit_cast(v16bf, aq0p);
  v16bf aq1 = __builtin_bit_cast(v16bf, aq1p);

  float sc = __expf(temp[0]);    // = DIM_HEAD^-0.5 = 0.125

  float m_run[8], l_run[8];
  v8f oacc[4];
  v8f zero = {};
#pragma unroll
  for (int t = 0; t < 4; ++t) oacc[t] = zero;
#pragma unroll
  for (int r = 0; r < 8; ++r) { m_run[r] = -3.3895e38f; l_run[r] = 0.f; }

  // TDM prologue: stage tile 0 (K and V) into buffer 0
  tdm_load_tile_32x64(lds_off(&Kt[0][0]), kg);
  tdm_load_tile_32x64(lds_off(&Vt[0][0]), vg);

  for (int j0 = 0; j0 < S_LEN; j0 += 32) {
    int cur = (j0 >> 5) & 1;
    if (j0 + 32 < S_LEN) {
      // prefetch next tile into the other buffer, then wait for current:
      // tensor ops complete in order, so TENSORcnt <= 2 => tile ready
      tdm_load_tile_32x64(lds_off(&Kt[cur ^ 1][0]), kg + (size_t)(j0 + 32) * QKV_LD);
      tdm_load_tile_32x64(lds_off(&Vt[cur ^ 1][0]), vg + (size_t)(j0 + 32) * QKV_LD);
      asm volatile("s_wait_tensorcnt 0x2" ::: "memory");
    } else {
      asm volatile("s_wait_tensorcnt 0x0" ::: "memory");
    }
    const unsigned short* Kc = &Kt[cur][0];   // [key 0..31][dim 0..63]
    const unsigned short* Vc = &Vt[cur][0];

    // ---- scores for 32 keys (two 16-key N tiles) from LDS ----
    v8f s[2];
#pragma unroll
    for (int t = 0; t < 2; ++t) {
      const unsigned short* krow = Kc + (t * 16 + ln) * 64;
      v8u b0p, b1p;
#pragma unroll
      for (int v = 0; v < 8; ++v) {
        int Kb = (g << 4) + (v << 1);
        b0p[v] = *(const unsigned int*)(krow + Kb);
        b1p[v] = *(const unsigned int*)(krow + 32 + Kb);
      }
      v8f z = {};
      z = __builtin_amdgcn_wmma_f32_16x16x32_bf16(
          false, aq0, false, __builtin_bit_cast(v16bf, b0p), (short)0, z, false, false);
      z = __builtin_amdgcn_wmma_f32_16x16x32_bf16(
          false, aq1, false, __builtin_bit_cast(v16bf, b1p), (short)0, z, false, false);
      s[t] = z;
    }

    // ---- mask + online softmax (row m = r + 8g lives in half-wave g) ----
#pragma unroll
    for (int r = 0; r < 8; ++r) {
      int i = i0 + r + (g << 3);
      int ja = j0 + ln, jb = j0 + 16 + ln;
      float x0 = s[0][r] * sc;
      float x1 = s[1][r] * sc;
      if (i >= ja && (i - ja) <= 7) x0 = -3.3895e38f;   // ~ -FLT_MAX (band fill)
      if (i >= jb && (i - jb) <= 7) x1 = -3.3895e38f;
      float mx   = hmax16(fmaxf(x0, x1));
      float newM = fmaxf(m_run[r], mx);
      float rs   = __expf(m_run[r] - newM);
      m_run[r]   = newM;
      float e0   = __expf(x0 - newM);
      float e1   = __expf(x1 - newM);
      float ts   = hsum16(e0 + e1);
      l_run[r]   = l_run[r] * rs + ts;
      oacc[0][r] *= rs; oacc[1][r] *= rs; oacc[2][r] *= rs; oacc[3][r] *= rs;
      int mrow = r + (g << 3);
      Plds[mrow * 32 + ln]      = f2bf(e0);
      Plds[mrow * 32 + 16 + ln] = f2bf(e1);
    }
    __syncthreads();

    // ---- P: C-layout -> A-layout (16x32 bf16) via LDS ----
    v8u pap;
#pragma unroll
    for (int v = 0; v < 8; ++v) {
      int Ka = (g << 4) + (v << 1);
      pap[v] = *(const unsigned int*)&Plds[ln * 32 + Ka];
    }
    v16bf pa = __builtin_bit_cast(v16bf, pap);
    __syncthreads();

    // ---- O += P V : 4 N-tiles covering Dh=64, V from LDS ----
#pragma unroll
    for (int t = 0; t < 4; ++t) {
      v8u vbp;
#pragma unroll
      for (int v = 0; v < 8; ++v) {
        int kk = (g << 4) + (v << 1);
        const unsigned short* vr = Vc + kk * 64 + t * 16 + ln;
        unsigned int lo = vr[0];
        unsigned int hi = vr[64];
        vbp[v] = lo | (hi << 16);
      }
      oacc[t] = __builtin_amdgcn_wmma_f32_16x16x32_bf16(
          false, pa, false, __builtin_bit_cast(v16bf, vbp), (short)0, oacc[t], false, false);
    }
  }

  // ---- normalize and store (bf16, [B,S,H*Dh] layout) ----
#pragma unroll
  for (int r = 0; r < 8; ++r) {
    float inv = 1.0f / l_run[r];
    int i = i0 + r + (g << 3);
    size_t orow = ((size_t)b * S_LEN + i) * INNER + h * 64;
    attn[orow + 0  + ln] = f2bf(oacc[0][r] * inv);
    attn[orow + 16 + ln] = f2bf(oacc[1][r] * inv);
    attn[orow + 32 + ln] = f2bf(oacc[2][r] * inv);
    attn[orow + 48 + ln] = f2bf(oacc[3][r] * inv);
  }
}

// ---------------------------------------------------------------------------
extern "C" void kernel_launch(void* const* d_in, const int* in_sizes, int n_in,
                              void* d_out, int out_size, void* d_ws, size_t ws_size,
                              hipStream_t stream) {
  (void)in_sizes; (void)n_in; (void)out_size; (void)ws_size;
  const float* x           = (const float*)d_in[0];
  const float* W_qkv       = (const float*)d_in[1];
  const float* u_qkv       = (const float*)d_in[2];
  const float* sigma_qkv   = (const float*)d_in[3];
  const float* W_out       = (const float*)d_in[4];
  const float* b_out       = (const float*)d_in[5];
  const float* u_out       = (const float*)d_in[6];
  const float* sigma_out   = (const float*)d_in[7];
  const float* temperature = (const float*)d_in[8];
  float* out = (float*)d_out;

  // workspace: [scale_qkv, scale_out] | qkv bf16 [4096 x 1536] | attn bf16 [4096 x 512]
  float* scale_qkv = (float*)d_ws;
  float* scale_out = scale_qkv + 1;
  unsigned short* qkvb  = (unsigned short*)((char*)d_ws + 256);
  unsigned short* attnb = qkvb + (size_t)4096 * QKV_LD;

  sn_sigma<<<1, 256, 0, stream>>>(W_qkv, u_qkv, sigma_qkv, scale_qkv, 1536, 512);
  sn_sigma<<<1, 256, 0, stream>>>(W_out, u_out, sigma_out, scale_out, 512, 512);

  // qkv = (x @ Wqkv^T) * scale  : M=4096, N=1536, K=512
  gemm_f32_bf16out<<<dim3(96, 256), 32, 0, stream>>>(x, W_qkv, scale_qkv, qkvb, 512, QKV_LD);

  // attention: 128 query tiles x (B*H)=16
  lsa_attn<<<dim3(128, 16), 32, 0, stream>>>(qkvb, temperature, attnb);

  // out = (attn @ Wout^T) * scale + b : M=4096, N=512, K=512
  gemm_bf16_f32out<<<dim3(32, 256), 32, 0, stream>>>(attnb, W_out, b_out, scale_out, out, 512, INNER);
}
